// BiAttentionLayer_23648089931911
// MI455X (gfx1250) — matrix operand order repaired and checked by reading, not compiled, not run on hardware
//
#include <hip/hip_runtime.h>
#include <hip/hip_bf16.h>

#define TT 4096
#define DD 512
#define LOG2E 1.4426950408889634f

typedef __attribute__((ext_vector_type(16))) __bf16 v16bf;
typedef __attribute__((ext_vector_type(8)))  float  v8f;
typedef __attribute__((ext_vector_type(4)))  unsigned int u32x4;
typedef __attribute__((ext_vector_type(4)))  float f32x4;

union BFrag { v16bf v; u32x4 u[2]; };

__device__ __forceinline__ unsigned short f2bf(float f) {
    unsigned u = __builtin_bit_cast(unsigned, f);
    u = (u + 0x7FFFu + ((u >> 16) & 1u)) >> 16;   // round-to-nearest-even
    return (unsigned short)u;
}

// ---------------------------------------------------------------------------
// prep: s2[r] = dot(question[r], w2)*log2(e); bf16 copies of ctx*w3*log2(e)
// (row-major), question (row-major) and question^T (for the PV B operand).
// Working in base-2 lets the softmax use v_exp_f32 (native exp2) directly.
// ---------------------------------------------------------------------------
__global__ void __launch_bounds__(256) prep_kernel(
    const float* __restrict__ x, const float* __restrict__ w,
    unsigned short* __restrict__ cw3bf, unsigned short* __restrict__ qbf,
    unsigned short* __restrict__ qTbf, float* __restrict__ s2)
{
    const int r = blockIdx.x;
    const int t = threadIdx.x;
    const float* ctx = x;
    const float* q   = x + TT * DD;
    __shared__ float red[256];

    float c0 = ctx[r * DD + t],        c1 = ctx[r * DD + 256 + t];
    float q0 = q[r * DD + t],          q1 = q[r * DD + 256 + t];
    float w3a = w[2 * DD + t],         w3b = w[2 * DD + 256 + t];

    cw3bf[r * DD + t]       = f2bf(c0 * w3a * LOG2E);
    cw3bf[r * DD + 256 + t] = f2bf(c1 * w3b * LOG2E);
    qbf[r * DD + t]         = f2bf(q0);
    qbf[r * DD + 256 + t]   = f2bf(q1);
    qTbf[t * TT + r]         = f2bf(q0);
    qTbf[(t + 256) * TT + r] = f2bf(q1);

    red[t] = q0 * w[DD + t] + q1 * w[DD + 256 + t];
    __syncthreads();
    for (int s = 128; s > 0; s >>= 1) {
        if (t < s) red[t] += red[t + s];
        __syncthreads();
    }
    if (t == 0) s2[r] = red[0] * LOG2E;
}

// ---------------------------------------------------------------------------
// attn: fused flash-attention. One WG = 16 context rows, 8 waves, 256 threads.
// Per 128-column block: wave wv computes S columns [wv*16, wv*16+16) (16
// WMMAs over K=512), waves exchange row maxima via LDS, P (bf16, base-2
// softmax numerator) goes to LDS, then each wave does P @ q for its 64-wide
// D-slice (16 WMMAs, K=128) plus a P @ ones WMMA that accumulates the row
// sums l in-register (same online rescale as the output accumulator).
// The per-row bias (ctx@w1) is softmax-invariant and dropped.
// b[i] = max_j A[i,j] = 1/l_i.
// ---------------------------------------------------------------------------
__global__ void __launch_bounds__(256)
__attribute__((amdgpu_waves_per_eu(4)))       // target <=256 VGPRs: 4 waves/SIMD
attn_kernel(
    const float* __restrict__ x,
    const unsigned short* __restrict__ cw3bf,
    const unsigned short* __restrict__ qbf,
    const unsigned short* __restrict__ qTbf,
    const float* __restrict__ s2,
    float* __restrict__ out, float* __restrict__ binv)
{
    const int lane = threadIdx.x & 31;
    const int wv   = threadIdx.x >> 5;   // 0..7
    const int half = lane >> 4;          // 0/1
    const int lx   = lane & 15;
    const int i0   = blockIdx.x * 16;

    __shared__ __align__(16) unsigned short Pl[16 * 136]; // 16x128 P, pad->136
    __shared__ __align__(16) float tmax[16][8];           // [row][wave]

    // A fragments (cw3 rows i0..i0+15). ISA 16-bit A layout (16x32):
    // lanes<16: M=lane, K {0..7,16..23}; lanes>=16: M=lane-16, K {8..15,24..31}
    // -> two contiguous 16B loads. Hoist only K=0..127 (32 VGPRs); the rest is
    // re-streamed each iteration from the WGP$-resident 16KB panel through a
    // laundered pointer so LICM cannot hoist it back into registers.
    const unsigned short* abase = cw3bf + (i0 + lx) * DD;
    BFrag a1[4];
    #pragma unroll
    for (int kk = 0; kk < 4; ++kk) {
        const int k0 = kk * 32 + half * 8;
        a1[kk].u[0] = *(const u32x4*)(abase + k0);
        a1[kk].u[1] = *(const u32x4*)(abase + k0 + 16);
    }

    const v8f vz = (v8f){0.f,0.f,0.f,0.f,0.f,0.f,0.f,0.f};
    v8f acc[4];
    #pragma unroll
    for (int t = 0; t < 4; ++t) acc[t] = vz;
    v8f accl = vz;                        // row-sum accumulator (l)

    float m[8];
    #pragma unroll
    for (int r = 0; r < 8; ++r) m[r] = -INFINITY;

    BFrag ones;                           // bf16 1.0 B operand for row sums
    ones.u[0] = (u32x4){0x3F803F80u, 0x3F803F80u, 0x3F803F80u, 0x3F803F80u};
    ones.u[1] = ones.u[0];

    for (int jb = 0; jb < TT / 128; ++jb) {
        __syncthreads();                  // protect Pl / tmax reuse
        const int j0 = jb * 128 + wv * 16;
        const unsigned short* bbase = qbf + (j0 + lx) * DD;

        // Opaque copy of the A-panel base: blocks loop-invariant hoisting of
        // the streamed A-fragment loads (they stay normal, waitcnt-tracked
        // global_load_b128s, reloaded per iteration from L0).
        const unsigned short* ab = abase;
        asm volatile("" : "+v"(ab));

        if (jb + 1 < TT / 128)            // warm next block (global_prefetch_b8)
            __builtin_prefetch(qbf + (j0 + 128 + lx) * DD, 0, 1);

        // ---- GEMM1: S tile 16x16, K=512. Two independent accumulation
        // chains + batched loads so the scheduler can pipeline. ----
        v8f sA = vz, sB = vz;
        #pragma unroll
        for (int g = 0; g < 4; ++g) {
            BFrag b[4], a[4];
            #pragma unroll
            for (int u = 0; u < 4; ++u) {
                const int kk = g * 4 + u;
                const unsigned short* bp = bbase + kk * 32 + half * 8;
                b[u].u[0] = *(const u32x4*)(bp);
                b[u].u[1] = *(const u32x4*)(bp + 16);
                if (kk >= 4) {            // streamed A chunk (L0-hot)
                    const unsigned short* ap = ab + kk * 32 + half * 8;
                    a[u].u[0] = *(const u32x4*)(ap);
                    a[u].u[1] = *(const u32x4*)(ap + 16);
                } else {
                    a[u] = a1[kk];
                }
            }
            #pragma unroll
            for (int u = 0; u < 4; ++u) {
                if (u & 1)
                    sB = __builtin_amdgcn_wmma_f32_16x16x32_bf16(
                             false, a[u].v, false, b[u].v, (short)0, sB, false, false);
                else
                    sA = __builtin_amdgcn_wmma_f32_16x16x32_bf16(
                             false, a[u].v, false, b[u].v, (short)0, sA, false, false);
            }
        }
        v8f s = sA + sB;
        s += s2[j0 + lx];                 // per-column bias (base-2 units)

        // ---- tile row max: reduce across the 16 lanes of each half ----
        float tm[8];
        #pragma unroll
        for (int r = 0; r < 8; ++r) {
            float v = s[r];
            v = fmaxf(v, __shfl_xor(v, 1, 32));
            v = fmaxf(v, __shfl_xor(v, 2, 32));
            v = fmaxf(v, __shfl_xor(v, 4, 32));
            v = fmaxf(v, __shfl_xor(v, 8, 32));
            tm[r] = v;
        }
        if (lx == 0) {                    // lanes 0 & 16 publish 8 rows each
            #pragma unroll
            for (int r = 0; r < 8; ++r) tmax[half * 8 + r][wv] = tm[r];
        }
        __syncthreads();

        // ---- online softmax (base 2): new max, rescale factor, P -> LDS ----
        float sc[8];
        #pragma unroll
        for (int r = 0; r < 8; ++r) {
            const int row = half * 8 + r;
            f32x4 t0 = *(const f32x4*)&tmax[row][0];
            f32x4 t1 = *(const f32x4*)&tmax[row][4];
            float bm = fmaxf(fmaxf(t0[0], t0[1]), fmaxf(t0[2], t0[3]));
            bm = fmaxf(bm, fmaxf(fmaxf(t1[0], t1[1]), fmaxf(t1[2], t1[3])));
            const float nm = fmaxf(m[r], bm);
            sc[r] = __builtin_exp2f(m[r] - nm);
            m[r]  = nm;
            const float p = __builtin_exp2f(s[r] - nm);
            Pl[row * 136 + wv * 16 + lx] = f2bf(p);
        }
        __syncthreads();

        // ---- issue P-fragment LDS loads first so ds latency overlaps the
        // rescale VALU block, then GEMM2: O += P(16x128) @ q(128x64) and
        // l += P @ ones (row sums, same layout, same rescale) ----
        BFrag a2[4];                      // P fragments, 4 K-chunks of 32
        #pragma unroll
        for (int kc = 0; kc < 4; ++kc) {
            const unsigned short* ap = Pl + lx * 136 + kc * 32 + half * 8;
            a2[kc].u[0] = *(const u32x4*)(ap);
            a2[kc].u[1] = *(const u32x4*)(ap + 16);
        }
        #pragma unroll
        for (int t = 0; t < 4; ++t) {
            #pragma unroll
            for (int r = 0; r < 8; ++r) acc[t][r] *= sc[r];
        }
        #pragma unroll
        for (int r = 0; r < 8; ++r) accl[r] *= sc[r];

        #pragma unroll
        for (int kc = 0; kc < 4; ++kc) {
            #pragma unroll
            for (int t = 0; t < 4; ++t) {
                BFrag b;                  // q^T: lane = fixed d-row, contiguous j
                const unsigned short* bp =
                    qTbf + (wv * 64 + t * 16 + lx) * TT + jb * 128 + kc * 32 + half * 8;
                b.u[0] = *(const u32x4*)(bp);
                b.u[1] = *(const u32x4*)(bp + 16);
                acc[t] = __builtin_amdgcn_wmma_f32_16x16x32_bf16(
                             false, a2[kc].v, false, b.v, (short)0, acc[t], false, false);
            }
            accl = __builtin_amdgcn_wmma_f32_16x16x32_bf16(
                       false, a2[kc].v, false, ones.v, (short)0, accl, false, false);
        }
    }

    // ---- epilogue: U = O/l; G[:,0:512]=ctx, [512:1024]=U, [1024:1536]=ctx*U
    #pragma unroll
    for (int t = 0; t < 4; ++t) {
        const int d = wv * 64 + t * 16 + lx;
        #pragma unroll
        for (int r = 0; r < 8; ++r) {
            const int i = i0 + half * 8 + r;
            const float u = acc[t][r] / accl[r];
            const float c = x[i * DD + d];
            out[i * 2048 + d]        = c;
            out[i * 2048 + 512 + d]  = u;
            out[i * 2048 + 1024 + d] = c * u;
        }
    }
    if (wv == 0 && lx == 0) {             // b[i] = max_j A[i,j] = 1/l_i
        #pragma unroll
        for (int r = 0; r < 8; ++r) binv[i0 + half * 8 + r] = 1.f / accl[r];
    }
}

// ---------------------------------------------------------------------------
// tail: h[d] = sum_i (1/l_i)*ctx[i][d];  G[:,1536:2048] = ctx*h  (no atomics)
// ---------------------------------------------------------------------------
__global__ void __launch_bounds__(256) tail_kernel(
    const float* __restrict__ x, const float* __restrict__ binv,
    float* __restrict__ out)
{
    const int d = blockIdx.x * 256 + threadIdx.x;  // 0..511
    const float* ctx = x;
    float h = 0.f;
    for (int i = 0; i < TT; ++i) h += binv[i] * ctx[i * DD + d];
    for (int i = 0; i < TT; ++i) out[i * 2048 + 1536 + d] = ctx[i * DD + d] * h;
}

extern "C" void kernel_launch(void* const* d_in, const int* in_sizes, int n_in,
                              void* d_out, int out_size, void* d_ws, size_t ws_size,
                              hipStream_t stream) {
    (void)in_sizes; (void)n_in; (void)out_size; (void)ws_size;
    const float* x = (const float*)d_in[0];
    const float* w = (const float*)d_in[1];
    float* out = (float*)d_out;
    char* ws = (char*)d_ws;

    unsigned short* cw3bf = (unsigned short*)(ws);                 // 4 MB
    unsigned short* qbf   = (unsigned short*)(ws + (4u << 20));    // 4 MB
    unsigned short* qTbf  = (unsigned short*)(ws + (8u << 20));    // 4 MB
    float*          s2    = (float*)(ws + (12u << 20));            // 16 KB
    float*          binv  = (float*)(ws + (12u << 20) + 16384);    // 16 KB

    prep_kernel<<<TT, 256, 0, stream>>>(x, w, cw3bf, qbf, qTbf, s2);
    attn_kernel<<<TT / 16, 256, 0, stream>>>(x, cw3bf, qbf, qTbf, s2, out, binv);
    tail_kernel<<<2, 256, 0, stream>>>(x, binv, out);
}
